// Restormer_cross_38860864094366
// MI455X (gfx1250) — matrix operand order repaired and checked by reading, not compiled, not run on hardware
//
#include <hip/hip_runtime.h>
#include <hip/hip_bf16.h>
#include <math.h>

#define DEVFN __device__ __forceinline__

typedef __attribute__((ext_vector_type(16))) __bf16 v16bf;
typedef __attribute__((ext_vector_type(8)))  __bf16 v8bf;
typedef __attribute__((ext_vector_type(4)))  __bf16 v4bf;
typedef __attribute__((ext_vector_type(8)))  float  v8f;

static constexpr int       BATCH = 4;
static constexpr int       DIMC  = 96;
static constexpr int       HEADS = 4;
static constexpr int       CH    = DIMC / HEADS;   // 24
static constexpr int       IMH   = 256, IMW = 256;
static constexpr long long NPIX  = (long long)IMH * IMW;   // 65536
static constexpr int       KSLICES = 8;
static constexpr long long SLICE   = NPIX / KSLICES;       // 8192
static constexpr int       GELEMS  = BATCH * DIMC * DIMC;  // 36864
static constexpr int       VCH   = 2 * DIMC;               // 192 v-channels

// ---- fp32 <-> bf16 helpers ---------------------------------------------
DEVFN __bf16 f2bf(float f) {                       // round to nearest even
  unsigned u = __builtin_bit_cast(unsigned, f);
  u += 0x7FFFu + ((u >> 16) & 1u);
  unsigned short h = (unsigned short)(u >> 16);
  return __builtin_bit_cast(__bf16, h);
}
DEVFN float bf2f(__bf16 h) {
  unsigned u = (unsigned)__builtin_bit_cast(unsigned short, h) << 16;
  return __builtin_bit_cast(float, u);
}

// ---- A tile 16x32 (MxK) from bf16 row-major (2 x b128 per lane) --------
// lane<16: M=lane, K in {kh..kh+7} U {16+kh..16+kh+7}, kh=0; lane>=16: kh=8.
DEVFN v16bf load_a_bf16(const __bf16* rowbase, int lane) {
  const int kh = (lane < 16) ? 0 : 8;
  v8bf c0 = *reinterpret_cast<const v8bf*>(rowbase + kh);
  v8bf c1 = *reinterpret_cast<const v8bf*>(rowbase + 16 + kh);
  v16bf a;
#pragma unroll
  for (int i = 0; i < 8; ++i) { a[i] = c0[i]; a[8 + i] = c1[i]; }
  return a;
}

// ---- B tile 32x16 (KxN), K contiguous per lane (2 x b128) --------------
// lane<16: N=lane, K in {0..7} U {16..23}; lane>=16: N=lane-16, K in {8..15} U {24..31}.
DEVFN v16bf load_b_bf16_contig(const __bf16* colbase, int lane) {
  const int ko = (lane < 16) ? 0 : 8;
  v8bf c0 = *reinterpret_cast<const v8bf*>(colbase + ko);
  v8bf c1 = *reinterpret_cast<const v8bf*>(colbase + ko + 16);
  v16bf b;
#pragma unroll
  for (int i = 0; i < 8; ++i) { b[i] = c0[i]; b[8 + i] = c1[i]; }
  return b;
}

// ========================================================================
// fp32 -> bf16 bulk convert (weights; layout preserved)
// ========================================================================
__global__ void cvt_f32_to_bf16_kernel(const float* __restrict__ src,
                                       __bf16* __restrict__ dst, long long n) {
  long long i = ((long long)blockIdx.x * blockDim.x + threadIdx.x) * 4;
  if (i + 3 < n) {
    float4 x = *reinterpret_cast<const float4*>(src + i);
    v4bf o; o[0] = f2bf(x.x); o[1] = f2bf(x.y); o[2] = f2bf(x.z); o[3] = f2bf(x.w);
    *reinterpret_cast<v4bf*>(dst + i) = o;
  } else {
    for (; i < n; ++i) dst[i] = f2bf(src[i]);
  }
}

// ========================================================================
// [B][96][N] fp32 (channels-first) -> [B][N][96] bf16 (channels-last)
// LDS-tiled 32x32 transpose; coalesced on both sides.
// ========================================================================
__global__ void transpose_cvt_kernel(const float* __restrict__ src,
                                     __bf16* __restrict__ dst) {
  __shared__ float t[32][33];
  const int b  = blockIdx.z;
  const int c0 = blockIdx.y * 32;
  const long long n0 = (long long)blockIdx.x * 32;
  const int tx = threadIdx.x & 31;
  const int ty = threadIdx.x >> 5;             // 0..7
  const float* sb = src + (long long)b * DIMC * NPIX;
#pragma unroll
  for (int i = 0; i < 4; ++i) {
    int row = ty + i * 8;                      // channel within tile
    t[row][tx] = sb[(long long)(c0 + row) * NPIX + n0 + tx];
  }
  __syncthreads();
  __bf16* db = dst + (long long)b * NPIX * DIMC;
#pragma unroll
  for (int i = 0; i < 4; ++i) {
    int row = ty + i * 8;                      // pixel within tile
    db[(n0 + row) * DIMC + c0 + tx] = f2bf(t[tx][row]);
  }
}

// ========================================================================
// conv1x1 as GEMM, channels-last activations:
//   Y[b,n,m] = sum_k W[(b),m,k] * X[b,n,xoff+k] + bias[m]
// All-b128 operand loads. OutT=bf16 -> channels-last b128 store;
// OutT=float -> channels-first store (d_out layout).
// One wave: 16 out-ch x 16 pixels, K=96 (3 WMMAs). 8 waves/block.
// ========================================================================
template <typename OutT>
__global__ void conv1x1_cl_wmma_kernel(const __bf16* __restrict__ X, long long x_bstride,
                                       int ldx, int xoff,
                                       const __bf16* __restrict__ Wt, long long w_bstride,
                                       const float* __restrict__ bias,
                                       OutT* __restrict__ Y, long long y_bstride, int ldy) {
  const int lane = threadIdx.x & 31;
  const int wave = threadIdx.x >> 5;
  const int b  = blockIdx.z;
  const int m0 = blockIdx.y * 16;
  const long long n0 = ((long long)blockIdx.x * 8 + wave) * 16;

  const __bf16* Xb = X + (long long)b * x_bstride;
  const __bf16* Wb = Wt + (long long)b * w_bstride;
  const int       m = m0 + (lane & 15);
  const long long n = n0 + (lane & 15);
  const __bf16* xcol = Xb + n * ldx + xoff;

  v8f acc = {};
#pragma unroll
  for (int kb = 0; kb < DIMC; kb += 32) {
    v16bf a  = load_a_bf16(Wb + (long long)m * DIMC + kb, lane);
    v16bf bt = load_b_bf16_contig(xcol + kb, lane);
    acc = __builtin_amdgcn_wmma_f32_16x16x32_bf16(false, a, false, bt, (short)0,
                                                  acc, false, false);
  }
  if constexpr (sizeof(OutT) == 2) {
    __bf16* Yb = (__bf16*)Y + (long long)b * y_bstride;
    const int mbase = m0 + ((lane < 16) ? 0 : 8);
    v8bf out;
#pragma unroll
    for (int r = 0; r < 8; ++r) out[r] = f2bf(acc[r] + bias[mbase + r]);
    *reinterpret_cast<v8bf*>(Yb + n * ldy + mbase) = out;   // b128 store
  } else {
    float* Yb = (float*)Y + (long long)b * y_bstride;
#pragma unroll
    for (int r = 0; r < 8; ++r) {
      int mr = m0 + ((lane < 16) ? r : (r + 8));
      Yb[(long long)mr * NPIX + n] = acc[r] + bias[mr];
    }
  }
}

// ========================================================================
// Depthwise 3x3 (SAME zero pad), channels-last in -> channels-FIRST out.
// Block: 32 pixels x 96 out-channels; LDS tile fuses the transpose.
// ========================================================================
__global__ void dwconv_cl2cf_kernel(const __bf16* __restrict__ X, int Cin, int cbase,
                                    const float* __restrict__ W9, const float* __restrict__ bias,
                                    __bf16* __restrict__ Y) {
  __shared__ float t[96][33];
  const int b = blockIdx.z;
  const long long n0 = (long long)blockIdx.x * 32;
  const __bf16* Xb = X + (long long)b * NPIX * Cin;
  for (int e = threadIdx.x; e < 96 * 32; e += 256) {
    int c  = e % 96;                 // coalesced over c (channels-last)
    int pl = e / 96;
    long long n = n0 + pl;
    int h = (int)(n / IMW), w = (int)(n % IMW);
    int ca = cbase + c;
    const float* wp = W9 + ca * 9;
    float s = bias[ca];
#pragma unroll
    for (int dy = -1; dy <= 1; ++dy) {
      int hh = h + dy;
      if (hh < 0 || hh >= IMH) continue;
#pragma unroll
      for (int dx = -1; dx <= 1; ++dx) {
        int ww = w + dx;
        if (ww < 0 || ww >= IMW) continue;
        s += wp[(dy + 1) * 3 + (dx + 1)] *
             bf2f(Xb[((long long)hh * IMW + ww) * Cin + ca]);
      }
    }
    t[c][pl] = s;
  }
  __syncthreads();
  __bf16* Yb = Y + (long long)b * DIMC * NPIX;
  for (int e = threadIdx.x; e < 96 * 32; e += 256) {
    int pl = e % 32;                 // coalesced over pixels (channels-first)
    int c  = e / 32;
    Yb[(long long)c * NPIX + n0 + pl] = f2bf(t[c][pl]);
  }
}

// ========================================================================
// Depthwise 3x3 for v channels (96..287 of kv), channels-last in/out.
// Output: [B][N][192] bf16 (v_scale = 0..95, v_shift = 96..191).
// ========================================================================
__global__ void dwconv_cl2cl_kernel(const __bf16* __restrict__ X,
                                    const float* __restrict__ W9, const float* __restrict__ bias,
                                    __bf16* __restrict__ Y) {
  long long idx = (long long)blockIdx.x * blockDim.x + threadIdx.x;
  long long total = (long long)BATCH * NPIX * VCH;
  if (idx >= total) return;
  int c = (int)(idx % VCH);                    // coalesced over c
  long long n = (idx / VCH) % NPIX;
  int b = (int)(idx / (VCH * NPIX));
  int ca = DIMC + c;                           // absolute kv channel
  int h = (int)(n / IMW), w = (int)(n % IMW);
  const __bf16* Xb = X + (long long)b * NPIX * (3 * DIMC);
  const float* wp = W9 + ca * 9;
  float s = bias[ca];
#pragma unroll
  for (int dy = -1; dy <= 1; ++dy) {
    int hh = h + dy;
    if (hh < 0 || hh >= IMH) continue;
#pragma unroll
    for (int dx = -1; dx <= 1; ++dx) {
      int ww = w + dx;
      if (ww < 0 || ww >= IMW) continue;
      s += wp[(dy + 1) * 3 + (dx + 1)] *
           bf2f(Xb[((long long)hh * IMW + ww) * (3 * DIMC) + ca]);
    }
  }
  Y[idx] = f2bf(s);
}

// ========================================================================
// Gram: Gpart[slice][b][m][n] = sum_{k in slice} q[b,m,k] * kmat[b,n,k]
// Channels-first bf16 rows; pure b128 + WMMA; deterministic split-K.
// ========================================================================
__global__ void gram_wmma_kernel(const __bf16* __restrict__ Q,
                                 const __bf16* __restrict__ Km,
                                 float* __restrict__ Gpart) {
  const int lane = threadIdx.x & 31;
  const int b  = blockIdx.z;
  const int m0 = (blockIdx.y / (DIMC / 16)) * 16;
  const int n0 = (blockIdx.y % (DIMC / 16)) * 16;
  const long long k0 = (long long)blockIdx.x * SLICE;

  const __bf16* Qrow = Q  + ((long long)b * DIMC + m0 + (lane & 15)) * NPIX;
  const __bf16* Kcol = Km + ((long long)b * DIMC + n0 + (lane & 15)) * NPIX;

  v8f acc = {};
  for (long long kk = k0; kk < k0 + SLICE; kk += 32) {
    v16bf a  = load_a_bf16(Qrow + kk, lane);
    v16bf bt = load_b_bf16_contig(Kcol + kk, lane);
    acc = __builtin_amdgcn_wmma_f32_16x16x32_bf16(false, a, false, bt, (short)0,
                                                  acc, false, false);
  }
  float* Gp = Gpart + (long long)blockIdx.x * GELEMS + (long long)b * DIMC * DIMC;
#pragma unroll
  for (int r = 0; r < 8; ++r) {
    int mr = m0 + ((lane < 16) ? r : (r + 8));
    int nr = n0 + (lane & 15);
    Gp[mr * DIMC + nr] = acc[r];
  }
}

__global__ void gram_reduce_kernel(const float* __restrict__ Gpart, float* __restrict__ G) {
  int i = blockIdx.x * blockDim.x + threadIdx.x;
  if (i >= GELEMS) return;
  float s = 0.f;
#pragma unroll
  for (int sl = 0; sl < KSLICES; ++sl) s += Gpart[(long long)sl * GELEMS + i];
  G[i] = s;
}

// ========================================================================
// Per-row inverse L2 norm over bf16 channels-first rows.
// ========================================================================
__global__ void rownorm_kernel(const __bf16* __restrict__ X, float* __restrict__ inv) {
  int row = blockIdx.x;                        // 0 .. BATCH*DIMC-1
  const __bf16* p = X + (long long)row * NPIX;
  float s = 0.f;
  for (long long i = threadIdx.x; i < NPIX; i += blockDim.x) {
    float v = bf2f(p[i]); s += v * v;
  }
  __shared__ float red[256];
  red[threadIdx.x] = s;
  __syncthreads();
  for (int o = 128; o > 0; o >>= 1) {
    if ((int)threadIdx.x < o) red[threadIdx.x] += red[threadIdx.x + o];
    __syncthreads();
  }
  if (threadIdx.x == 0) inv[row] = 1.0f / fmaxf(sqrtf(red[0]), 1e-12f);
}

// ========================================================================
// Softmax over d within each head block, folding in norms + temperature.
// ========================================================================
__global__ void attn_softmax_kernel(const float* __restrict__ G, const float* __restrict__ invq,
                                    const float* __restrict__ invk, const float* __restrict__ temp,
                                    float* __restrict__ attn) {
  int bh = blockIdx.x; int b = bh / HEADS; int h = bh % HEADS;
  int c = threadIdx.x;
  if (c >= CH) return;
  float t  = temp[h];
  int   cg = h * CH + c;
  float iq = invq[b * DIMC + cg];
  float row[CH];
  float mx = -1e30f;
  for (int d = 0; d < CH; ++d) {
    int dg = h * CH + d;
    float v = G[(b * DIMC + cg) * DIMC + dg] * iq * invk[b * DIMC + dg] * t;
    row[d] = v; mx = fmaxf(mx, v);
  }
  float sum = 0.f;
  for (int d = 0; d < CH; ++d) { row[d] = __expf(row[d] - mx); sum += row[d]; }
  float rs = 1.0f / sum;
  float* ap = attn + ((b * HEADS + h) * CH + c) * CH;
  for (int d = 0; d < CH; ++d) ap[d] = row[d] * rs;
}

// ========================================================================
// M_eff[b,o,dg] = sum_c ps_w[o, h*24+c] * attn[b,h,c,dg%24]   (h = dg/24)
// bf16 out: consumed only as WMMA A-matrix.
// ========================================================================
__global__ void build_meff_kernel(const float* __restrict__ attn, const float* __restrict__ psw,
                                  const float* __restrict__ pshw, __bf16* __restrict__ ms,
                                  __bf16* __restrict__ mh) {
  int idx = blockIdx.x * blockDim.x + threadIdx.x;
  if (idx >= GELEMS) return;
  int d = idx % DIMC;
  int o = (idx / DIMC) % DIMC;
  int b = idx / (DIMC * DIMC);
  int h = d / CH, dl = d % CH;
  const float* ap = attn + ((b * HEADS + h) * CH) * CH + dl;
  float ss = 0.f, sh = 0.f;
  for (int cl = 0; cl < CH; ++cl) {
    float a = ap[cl * CH];
    ss += psw [o * DIMC + h * CH + cl] * a;
    sh += pshw[o * DIMC + h * CH + cl] * a;
  }
  ms[idx] = f2bf(ss); mh[idx] = f2bf(sh);
}

// ========================================================================
extern "C" void kernel_launch(void* const* d_in, const int* in_sizes, int n_in,
                              void* d_out, int out_size, void* d_ws, size_t ws_size,
                              hipStream_t stream) {
  (void)in_sizes; (void)n_in; (void)out_size; (void)ws_size;
  const float* content = (const float*)d_in[0];
  const float* style   = (const float*)d_in[1];
  const float* temp    = (const float*)d_in[2];
  const float* q_w     = (const float*)d_in[3];
  const float* q_b     = (const float*)d_in[4];
  const float* qd_w    = (const float*)d_in[5];
  const float* qd_b    = (const float*)d_in[6];
  const float* kv_w    = (const float*)d_in[7];
  const float* kv_b    = (const float*)d_in[8];
  const float* kvd_w   = (const float*)d_in[9];
  const float* kvd_b   = (const float*)d_in[10];
  const float* ps_w    = (const float*)d_in[11];
  const float* ps_b    = (const float*)d_in[12];
  const float* psh_w   = (const float*)d_in[13];
  const float* psh_b   = (const float*)d_in[14];

  // ---- workspace layout --------------------------------------------------
  char* ws = (char*)d_ws;
  size_t off = 0;
  auto carve = [&](long long bytes) -> char* {
    char* p = ws + off;
    off += (size_t)((bytes + 255) & ~255LL);
    return p;
  };
  const long long el96 = (long long)BATCH * DIMC * NPIX;    // 25,165,824
  __bf16* content_cl = (__bf16*)carve(el96 * 2);            // [B][N][96]
  __bf16* style_cl   = (__bf16*)carve(el96 * 2);            // [B][N][96]
  __bf16* qw_bf      = (__bf16*)carve((long long)DIMC * DIMC * 2);
  __bf16* kvw_bf     = (__bf16*)carve((long long)3 * DIMC * DIMC * 2);
  __bf16* q_pre_cl   = (__bf16*)carve(el96 * 2);            // [B][N][96]
  __bf16* kv_pre_cl  = (__bf16*)carve(3 * el96 * 2);        // [B][N][288]
  __bf16* q_cf       = (__bf16*)carve(el96 * 2);            // [B][96][N]
  __bf16* k_cf       = (__bf16*)carve(el96 * 2);            // [B][96][N]
  __bf16* v_cl       = (__bf16*)carve(2 * el96 * 2);        // [B][N][192]
  float*  Gpart      = (float*) carve((long long)KSLICES * GELEMS * sizeof(float));
  float*  G          = (float*) carve((long long)GELEMS * sizeof(float));
  float*  invq       = (float*) carve((long long)BATCH * DIMC * sizeof(float));
  float*  invk       = (float*) carve((long long)BATCH * DIMC * sizeof(float));
  float*  attn       = (float*) carve((long long)BATCH * HEADS * CH * CH * sizeof(float));
  __bf16* meff_s     = (__bf16*)carve((long long)GELEMS * 2);
  __bf16* meff_h     = (__bf16*)carve((long long)GELEMS * 2);

  const dim3 blk256(256);
  const int nxTiles = (int)(NPIX / (16 * 8));               // 512

  // 0) input transposes to channels-last bf16 + weight converts
  transpose_cvt_kernel<<<dim3((unsigned)(NPIX / 32), DIMC / 32, BATCH), blk256, 0, stream>>>(
      content, content_cl);
  transpose_cvt_kernel<<<dim3((unsigned)(NPIX / 32), DIMC / 32, BATCH), blk256, 0, stream>>>(
      style, style_cl);
  {
    long long nqw = (long long)DIMC * DIMC, nkv = 3 * nqw;
    cvt_f32_to_bf16_kernel<<<dim3((unsigned)((nqw / 4 + 255) / 256)), blk256, 0, stream>>>(q_w, qw_bf, nqw);
    cvt_f32_to_bf16_kernel<<<dim3((unsigned)((nkv / 4 + 255) / 256)), blk256, 0, stream>>>(kv_w, kvw_bf, nkv);
  }

  // 1) channel projections (channels-last GEMMs, all-b128 + WMMA)
  conv1x1_cl_wmma_kernel<__bf16><<<dim3(nxTiles, DIMC / 16, BATCH), blk256, 0, stream>>>(
      content_cl, NPIX * DIMC, DIMC, 0, qw_bf, 0, q_b, q_pre_cl, NPIX * DIMC, DIMC);
  conv1x1_cl_wmma_kernel<__bf16><<<dim3(nxTiles, (3 * DIMC) / 16, BATCH), blk256, 0, stream>>>(
      style_cl, NPIX * DIMC, DIMC, 0, kvw_bf, 0, kv_b, kv_pre_cl, NPIX * 3 * DIMC, 3 * DIMC);

  // 2) depthwise 3x3: q,k fused transpose to channels-first; v stays CL
  dwconv_cl2cf_kernel<<<dim3((unsigned)(NPIX / 32), 1, BATCH), blk256, 0, stream>>>(
      q_pre_cl, DIMC, 0, qd_w, qd_b, q_cf);
  dwconv_cl2cf_kernel<<<dim3((unsigned)(NPIX / 32), 1, BATCH), blk256, 0, stream>>>(
      kv_pre_cl, 3 * DIMC, 0, kvd_w, kvd_b, k_cf);
  {
    long long tv = (long long)BATCH * NPIX * VCH;
    dwconv_cl2cl_kernel<<<dim3((unsigned)((tv + 255) / 256)), blk256, 0, stream>>>(
        kv_pre_cl, kvd_w, kvd_b, v_cl);
  }

  // 3) Gram matrix G = q . k^T (split-K slices, deterministic reduce)
  gram_wmma_kernel<<<dim3(KSLICES, (DIMC / 16) * (DIMC / 16), BATCH), dim3(32), 0, stream>>>(
      q_cf, k_cf, Gpart);
  gram_reduce_kernel<<<dim3((GELEMS + 255) / 256), blk256, 0, stream>>>(Gpart, G);

  // 4) row norms of q and k
  rownorm_kernel<<<dim3(BATCH * DIMC), blk256, 0, stream>>>(q_cf, invq);
  rownorm_kernel<<<dim3(BATCH * DIMC), blk256, 0, stream>>>(k_cf, invk);

  // 5) softmax (norms + temperature folded in)
  attn_softmax_kernel<<<dim3(BATCH * HEADS), dim3(32), 0, stream>>>(G, invq, invk, temp, attn);

  // 6) effective output-projection weights M_eff = ps_w . blockdiag(attn)
  build_meff_kernel<<<dim3((GELEMS + 255) / 256), blk256, 0, stream>>>(
      attn, ps_w, psh_w, meff_s, meff_h);

  // 7) fused (attn @ v) + output conv1x1: CL bf16 in, CF fp32 out (d_out)
  float* out_scale = (float*)d_out;
  float* out_shift = out_scale + (long long)BATCH * DIMC * NPIX;
  conv1x1_cl_wmma_kernel<float><<<dim3(nxTiles, DIMC / 16, BATCH), blk256, 0, stream>>>(
      v_cl, NPIX * VCH, VCH, 0,    meff_s, (long long)DIMC * DIMC, ps_b,
      out_scale, (long long)DIMC * NPIX, 0);
  conv1x1_cl_wmma_kernel<float><<<dim3(nxTiles, DIMC / 16, BATCH), blk256, 0, stream>>>(
      v_cl, NPIX * VCH, VCH, DIMC, meff_h, (long long)DIMC * DIMC, psh_b,
      out_shift, (long long)DIMC * NPIX, 0);
}
